// CausalSelfAttention_88974542504036
// MI455X (gfx1250) — matrix operand order repaired and checked by reading, not compiled, not run on hardware
//
#include <hip/hip_runtime.h>
#include <hip/hip_bf16.h>
#include <math.h>

// ---------------------------------------------------------------------------
// Types
// ---------------------------------------------------------------------------
typedef _Float16 v16h __attribute__((ext_vector_type(16)));
typedef _Float16 v8h  __attribute__((ext_vector_type(8)));
typedef float    v8f  __attribute__((ext_vector_type(8)));

// Problem constants (match reference)
constexpr int B_  = 2;
constexpr int T_  = 2048;
constexpr int C_  = 2048;
constexpr int NH  = 16;
constexpr int NKV = 4;
constexpr int HD  = 128;             // head dim
constexpr int KVD = NKV * HD;        // 512
constexpr int QKV_N = C_ + 2 * KVD;  // 3072
constexpr int Tp  = T_ + 32;         // padded key length (tile overreach safety)

// ---------------------------------------------------------------------------
// WMMA helpers
// ---------------------------------------------------------------------------
__device__ inline v8f vzero8() {
    v8f z;
#pragma unroll
    for (int i = 0; i < 8; ++i) z[i] = 0.0f;
    return z;
}

__device__ inline v8f wmma_f16(v16h a, v16h b, v8f c) {
    // D = A(16x32 f16) * B(32x16 f16) + C(16x16 f32)
    return __builtin_amdgcn_wmma_f32_16x16x32_f16(
        /*neg_a=*/false, a, /*neg_b=*/false, b,
        /*c_mod=*/(short)0, c, /*reuse_a=*/false, /*reuse_b=*/false);
}

// A-fragment (16x32, f16), source row-major, K contiguous (stride ld).
// ISA layout: lane<16 holds row (lane&15), K {0..7,16..23}; lane>=16 holds
// same row, K {8..15,24..31}; v16h elems 0..7 = first 8 K, 8..15 = +16.
__device__ inline v16h load_frag_a(const _Float16* base, int lane, size_t ld) {
    int r  = lane & 15;
    int hi = lane >> 4;
    const _Float16* p = base + (size_t)r * ld + (size_t)hi * 8;
    v8h lo  = *(const v8h*)(p);
    v8h hi8 = *(const v8h*)(p + 16);
    return __builtin_shufflevector(lo, hi8, 0,1,2,3,4,5,6,7,8,9,10,11,12,13,14,15);
}

// B-fragment (32x16, f16) for an NT source: source rows are B's N columns,
// source cols are the contraction K (contiguous, stride ld).
// Layout (matches A-fragment): lane<16 (col = lane&15) holds K {0..7,16..23};
// lane>=16 holds K {8..15,24..31}; elems 0..7 = first 8 K, 8..15 = +16.
__device__ inline v16h load_frag_b(const _Float16* base, int lane, size_t ld) {
    int r  = lane & 15;
    int hi = lane >> 4;
    const _Float16* p = base + (size_t)r * ld + (size_t)hi * 8;
    v8h a = *(const v8h*)(p);
    v8h b = *(const v8h*)(p + 16);
    return __builtin_shufflevector(a, b, 0,1,2,3,4,5,6,7,8,9,10,11,12,13,14,15);
}

// ---------------------------------------------------------------------------
// Kernel 1: fp32 -> f16 convert
// ---------------------------------------------------------------------------
__global__ void cvt_f32_f16(const float* __restrict__ in, _Float16* __restrict__ out, int n) {
    int i = blockIdx.x * blockDim.x + threadIdx.x;
    if (i < n) out[i] = (_Float16)in[i];
}

// ---------------------------------------------------------------------------
// Kernel 2/5: NT GEMM, f16 inputs, fp32 output.
// C[m][n] = sum_k A[m][k] * Bm[n][k].  Each wave computes a 64x64 tile with a
// 4x4 grid of 16x16 WMMA accumulators. blockDim = (32, 4).
// ---------------------------------------------------------------------------
__global__ __launch_bounds__(128)
void gemm_nt_f16(const _Float16* __restrict__ A, const _Float16* __restrict__ Bm,
                 float* __restrict__ C, int M, int N, int K) {
    int lane  = threadIdx.x;
    int mbase = (blockIdx.y * 4 + threadIdx.y) * 64;
    int nbase = blockIdx.x * 64;
    if (mbase >= M) return;

    v8f acc[4][4];
#pragma unroll
    for (int i = 0; i < 4; ++i)
#pragma unroll
        for (int j = 0; j < 4; ++j) acc[i][j] = vzero8();

    for (int kb = 0; kb < K; kb += 32) {
        v16h af[4], bf[4];
#pragma unroll
        for (int i = 0; i < 4; ++i)
            af[i] = load_frag_a(A + (size_t)(mbase + 16 * i) * K + kb, lane, K);
#pragma unroll
        for (int j = 0; j < 4; ++j)
            bf[j] = load_frag_b(Bm + (size_t)(nbase + 16 * j) * K + kb, lane, K);
#pragma unroll
        for (int i = 0; i < 4; ++i)
#pragma unroll
            for (int j = 0; j < 4; ++j)
                acc[i][j] = wmma_f16(af[i], bf[j], acc[i][j]);
    }

    int col = lane & 15;
    int r0  = (lane >> 4) * 8;
#pragma unroll
    for (int i = 0; i < 4; ++i)
#pragma unroll
        for (int j = 0; j < 4; ++j)
#pragma unroll
            for (int e = 0; e < 8; ++e)
                C[(size_t)(mbase + 16 * i + r0 + e) * N + (nbase + 16 * j + col)] = acc[i][j][e];
}

// ---------------------------------------------------------------------------
// Kernel 3: RoPE + RMS-norm on q/k, transpose v.
// One block per (b, t) over padded Tp; 128 threads = head_dim lanes.
//   q_h: (b*T+t, h, d) f16 row-major
//   k_h: (b*Tp+t, kvh, d) f16 row-major (zero-padded rows t>=T)
//   vt : (b, kvh, d, t) f16 (zero-padded cols t>=T)
// ---------------------------------------------------------------------------
__global__ __launch_bounds__(128)
void rope_rms_kernel(const float* __restrict__ qkv,
                     const float* __restrict__ cosp, const float* __restrict__ sinp,
                     _Float16* __restrict__ q_h, _Float16* __restrict__ k_h,
                     _Float16* __restrict__ vt) {
    int blk = blockIdx.x;          // 0 .. B_*Tp-1
    int b = blk / Tp;
    int t = blk % Tp;
    int d = threadIdx.x;           // 0..127

    if (t >= T_) {                 // zero the padding rows
#pragma unroll
        for (int kvh = 0; kvh < NKV; ++kvh) {
            k_h[((size_t)(b * Tp + t)) * KVD + kvh * HD + d] = (_Float16)0.0f;
            vt[(((size_t)(b * NKV + kvh)) * HD + d) * Tp + t] = (_Float16)0.0f;
        }
        return;
    }

    const float* row = qkv + (size_t)(b * T_ + t) * QKV_N;
    int j = d & 63;
    float cv = cosp[(size_t)t * 64 + j];
    float sv = sinp[(size_t)t * 64 + j];

    __shared__ float red[4];

    for (int hh = 0; hh < NH + NKV; ++hh) {
        const float* hp = (hh < NH) ? (row + hh * HD) : (row + C_ + (hh - NH) * HD);
        float x1 = hp[j];
        float x2 = hp[j + 64];
        float r  = (d < 64) ? (x1 * cv + x2 * sv) : (-x1 * sv + x2 * cv);

        float sq = r * r;
#pragma unroll
        for (int off = 1; off < 32; off <<= 1) sq += __shfl_xor(sq, off, 32);
        if ((threadIdx.x & 31) == 0) red[threadIdx.x >> 5] = sq;
        __syncthreads();
        float ms  = (red[0] + red[1] + red[2] + red[3]) * (1.0f / 128.0f);
        float inv = rsqrtf(ms + 1.1920929e-7f);
        __syncthreads();

        _Float16 out = (_Float16)(r * inv);
        if (hh < NH)
            q_h[(size_t)(b * T_ + t) * C_ + hh * HD + d] = out;
        else
            k_h[((size_t)(b * Tp + t)) * KVD + (hh - NH) * HD + d] = out;
    }

#pragma unroll
    for (int kvh = 0; kvh < NKV; ++kvh)
        vt[(((size_t)(b * NKV + kvh)) * HD + d) * Tp + t] =
            (_Float16)row[C_ + KVD + kvh * HD + d];
}

// ---------------------------------------------------------------------------
// Kernel 4: sliding-window flash attention. One wave per (qtile, head, batch).
// S = Q(16x128) Kt, online softmax, P transposed via LDS, O += P V.
// ---------------------------------------------------------------------------
__global__ __launch_bounds__(32)
void attn_kernel(const _Float16* __restrict__ q_h, const _Float16* __restrict__ k_h,
                 const _Float16* __restrict__ vt, _Float16* __restrict__ y_h,
                 const int* __restrict__ window_p) {
    int lane = threadIdx.x;
    int qb   = blockIdx.x * 16;
    int h    = blockIdx.y;
    int b    = blockIdx.z;
    int kvh  = h >> 2;                     // NH / NKV = 4
    int window = *window_p;

    const _Float16* qbase = q_h + (size_t)(b * T_ + qb) * C_ + h * HD;
    const _Float16* kbase = k_h + (size_t)b * Tp * KVD + kvh * HD;
    const _Float16* vbase = vt + ((size_t)(b * NKV + kvh)) * HD * Tp;

    v16h qf[4];
#pragma unroll
    for (int c = 0; c < 4; ++c)
        qf[c] = load_frag_a(qbase + c * 32, lane, C_);

    float Mrow[8], Lrow[8];
    v8f accO[8];
#pragma unroll
    for (int e = 0; e < 8; ++e) { Mrow[e] = -INFINITY; Lrow[e] = 0.0f; }
#pragma unroll
    for (int c = 0; c < 8; ++c) accO[c] = vzero8();

    __shared__ alignas(16) _Float16 Pl[16 * 32];

    const float scale = 0.08838834764831845f;   // 1/sqrt(128)
    int col = lane & 15;
    int hf  = lane >> 4;

    int klo = qb - window;
    int k0  = (klo <= 0) ? 0 : (klo & ~31);
    int kend = qb + 16;

    for (int kb = k0; kb < kend; kb += 32) {
        // ---- scores: S[16 x 32] = Q * K^T --------------------------------
        v8f s0 = vzero8(), s1 = vzero8();
#pragma unroll
        for (int c = 0; c < 4; ++c) {
            v16h kf0 = load_frag_b(kbase + (size_t)kb * KVD + c * 32, lane, KVD);
            v16h kf1 = load_frag_b(kbase + (size_t)(kb + 16) * KVD + c * 32, lane, KVD);
            s0 = wmma_f16(qf[c], kf0, s0);
            s1 = wmma_f16(qf[c], kf1, s1);
        }

        // ---- mask + online softmax ---------------------------------------
        int n0 = kb + col;
        int n1 = n0 + 16;
        float alpha[8];
#pragma unroll
        for (int e = 0; e < 8; ++e) {
            int m = qb + hf * 8 + e;
            float v0 = ((n0 <= m) && (n0 >= m - window)) ? s0[e] * scale : -INFINITY;
            float v1 = ((n1 <= m) && (n1 >= m - window)) ? s1[e] * scale : -INFINITY;
            float mx = fmaxf(v0, v1);
#pragma unroll
            for (int off = 1; off < 16; off <<= 1)
                mx = fmaxf(mx, __shfl_xor(mx, off, 32));
            float mn = fmaxf(Mrow[e], mx);
            float p0, p1, al;
            if (mn == -INFINITY) { al = 1.0f; p0 = 0.0f; p1 = 0.0f; }
            else {
                al = __expf(Mrow[e] - mn);
                p0 = __expf(v0 - mn);
                p1 = __expf(v1 - mn);
            }
            float rs = p0 + p1;
#pragma unroll
            for (int off = 1; off < 16; off <<= 1)
                rs += __shfl_xor(rs, off, 32);
            Lrow[e] = Lrow[e] * al + rs;
            Mrow[e] = mn;
            alpha[e] = al;
            // P row-major into LDS for the A-fragment reload
            Pl[(hf * 8 + e) * 32 + col]      = (_Float16)p0;
            Pl[(hf * 8 + e) * 32 + col + 16] = (_Float16)p1;
        }

#pragma unroll
        for (int c = 0; c < 8; ++c)
#pragma unroll
            for (int e = 0; e < 8; ++e) accO[c][e] *= alpha[e];

        // cross-lane LDS store -> load: wait for DS stores to land
        asm volatile("s_wait_dscnt 0" ::: "memory");
        v16h pf = load_frag_a(&Pl[0], lane, 32);

        // ---- O += P * V ---------------------------------------------------
#pragma unroll
        for (int c = 0; c < 8; ++c) {
            v16h vf = load_frag_b(vbase + (size_t)(c * 16) * Tp + kb, lane, Tp);
            accO[c] = wmma_f16(pf, vf, accO[c]);
        }
    }

    // epilogue: normalize and store to y (f16, row-major (b,t,h,d))
#pragma unroll
    for (int c = 0; c < 8; ++c)
#pragma unroll
        for (int e = 0; e < 8; ++e) {
            float val = accO[c][e] / Lrow[e];
            y_h[(size_t)(b * T_ + qb + hf * 8 + e) * C_ + h * HD + c * 16 + col] =
                (_Float16)val;
        }
}

// ---------------------------------------------------------------------------
// Host launcher
// ---------------------------------------------------------------------------
extern "C" void kernel_launch(void* const* d_in, const int* in_sizes, int n_in,
                              void* d_out, int out_size, void* d_ws, size_t ws_size,
                              hipStream_t stream) {
    const float* x    = (const float*)d_in[0];
    const float* cosp = (const float*)d_in[1];
    const float* sinp = (const float*)d_in[2];
    const float* Wq   = (const float*)d_in[3];
    const float* Wk   = (const float*)d_in[4];
    const float* Wv   = (const float*)d_in[5];
    const float* Wo   = (const float*)d_in[6];
    const int*   win  = (const int*)d_in[7];
    float* out = (float*)d_out;

    const int M = B_ * T_;   // 4096

    // workspace carve-up (256B aligned)
    size_t off = 0;
    auto carve = [&](size_t bytes) {
        size_t o = off;
        off = (off + bytes + 255) & ~(size_t)255;
        return o;
    };
    char* ws = (char*)d_ws;
    _Float16* xh    = (_Float16*)(ws + carve((size_t)M * C_ * 2));
    _Float16* wqkv  = (_Float16*)(ws + carve((size_t)QKV_N * C_ * 2));
    _Float16* wo_h  = (_Float16*)(ws + carve((size_t)C_ * C_ * 2));
    float*    qkvf  = (float*)   (ws + carve((size_t)M * QKV_N * 4));
    _Float16* q_h   = (_Float16*)(ws + carve((size_t)M * C_ * 2));
    _Float16* k_h   = (_Float16*)(ws + carve((size_t)B_ * Tp * KVD * 2));
    _Float16* vt    = (_Float16*)(ws + carve((size_t)B_ * NKV * HD * Tp * 2));
    _Float16* y_h   = (_Float16*)(ws + carve((size_t)M * C_ * 2));
    (void)ws_size; (void)n_in; (void)in_sizes; (void)out_size;

    // 1) f32 -> f16 converts (Wq/Wk/Wv packed as one 3072 x 2048 NT B-matrix)
    {
        int n;
        n = M * C_;
        cvt_f32_f16<<<(n + 255) / 256, 256, 0, stream>>>(x, xh, n);
        n = C_ * C_;
        cvt_f32_f16<<<(n + 255) / 256, 256, 0, stream>>>(Wq, wqkv, n);
        n = KVD * C_;
        cvt_f32_f16<<<(n + 255) / 256, 256, 0, stream>>>(Wk, wqkv + (size_t)C_ * C_, n);
        cvt_f32_f16<<<(n + 255) / 256, 256, 0, stream>>>(Wv, wqkv + (size_t)(C_ + KVD) * C_, n);
        n = C_ * C_;
        cvt_f32_f16<<<(n + 255) / 256, 256, 0, stream>>>(Wo, wo_h, n);
    }

    // 2) qkv = x @ [Wq;Wk;Wv]^T   (4096 x 3072 x 2048)
    {
        dim3 grid(QKV_N / 64, M / 256);
        dim3 block(32, 4);
        gemm_nt_f16<<<grid, block, 0, stream>>>(xh, wqkv, qkvf, M, QKV_N, C_);
    }

    // 3) RoPE + RMS-norm + V transpose
    rope_rms_kernel<<<B_ * Tp, 128, 0, stream>>>(qkvf, cosp, sinp, q_h, k_h, vt);

    // 4) sliding-window attention
    {
        dim3 grid(T_ / 16, NH, B_);
        attn_kernel<<<grid, 32, 0, stream>>>(q_h, k_h, vt, y_h, win);
    }

    // 5) out = y @ Wo^T   (4096 x 2048 x 2048)
    {
        dim3 grid(C_ / 64, M / 256);
        dim3 block(32, 4);
        gemm_nt_f16<<<grid, block, 0, stream>>>(y_h, wo_h, out, M, C_, C_);
    }
}
